// SynthesisLayer_32942399160553
// MI455X (gfx1250) — matrix operand order repaired and checked by reading, not compiled, not run on hardware
//
#include <hip/hip_runtime.h>
#include <cstdint>
#include <cstddef>

// ---------------------------------------------------------------------------
// StyleGAN2-style modulated upsample conv layer for MI455X (gfx1250).
// Main conv (~77 GFLOP vs ~200MB traffic -> compute bound) runs as a
// phase-decomposed implicit GEMM on v_wmma_f32_16x16x32_bf16, with
// cooperative LDS staging via GLOBAL_LOAD_ASYNC_TO_LDS (ASYNCcnt).
// A and B both use contiguous k (k = 16h + s), so every WMMA fragment is one contiguous
// 16-element LDS run (2 adjacent ds_load_b128, zero shuffle VALU).
// ---------------------------------------------------------------------------

typedef __attribute__((ext_vector_type(16))) __bf16   v16bf;
typedef __attribute__((ext_vector_type(8)))  __bf16   v8bf;
typedef __attribute__((ext_vector_type(8)))  float    v8f;
typedef __attribute__((ext_vector_type(8)))  uint16_t v8u16;
typedef __attribute__((ext_vector_type(4)))  int      v4i;

#define NB    16          // batch
#define CIN   512
#define COUT  512
#define RIN   32          // input spatial
#define ROUT  64          // output spatial
#define Y1DIM 65          // pre-blur spatial (transposed-conv output)
#define Y1PIX (Y1DIM*Y1DIM)

#define KCH   256         // K chunk staged in LDS per step
#define KPAD  (KCH + 8)   // +16B row pad: row stride = 132 dwords = 4 banks mod 64
#define LDS_BYTES ((128 + 64) * KPAD * 2)   // A[128][KPAD] + B[64][KPAD] bf16

#if defined(__has_builtin)
#if __has_builtin(__builtin_amdgcn_global_load_async_to_lds_b128)
#define HAVE_ASYNC_LDS 1
#endif
#endif
#ifndef HAVE_ASYNC_LDS
#define HAVE_ASYNC_LDS 0
#endif

typedef __attribute__((address_space(1))) v4i* gv4i_ptr;   // global int4*
typedef __attribute__((address_space(3))) v4i* lv4i_ptr;   // LDS    int4*

__device__ __forceinline__ uint16_t f32_to_bf16_bits(float f) {
  union { float f; uint32_t u; } a; a.f = f;
  uint32_t r = a.u + 0x7FFFu + ((a.u >> 16) & 1u);   // round-to-nearest-even
  return (uint16_t)(r >> 16);
}

// 16-byte global -> LDS stage (async DMA path; ASYNCcnt-tracked).
__device__ __forceinline__ void stage16(uint16_t* ldst, const uint16_t* gsrc) {
#if HAVE_ASYNC_LDS
  __builtin_amdgcn_global_load_async_to_lds_b128(
      (gv4i_ptr)gsrc, (lv4i_ptr)ldst, 0, 0);
#else
  *(v8u16*)ldst = *(const v8u16*)gsrc;
#endif
}

__device__ __forceinline__ void async_stage_fence() {
#if HAVE_ASYNC_LDS
#if __has_builtin(__builtin_amdgcn_s_wait_asynccnt)
  __builtin_amdgcn_s_wait_asynccnt(0);
#else
  asm volatile("s_wait_asynccnt 0x0" ::: "memory");
#endif
#endif
}

// One contiguous 16-element bf16 fragment: two adjacent 16B LDS loads,
// concatenated without element moves.
__device__ __forceinline__ v16bf make_frag16(const uint16_t* p) {
  v8bf lo = *(const v8bf*)p;
  v8bf hi = *(const v8bf*)(p + 8);
  return __builtin_shufflevector(lo, hi, 0, 1, 2, 3, 4, 5, 6, 7,
                                         8, 9, 10, 11, 12, 13, 14, 15);
}

// ---------------------------------------------------------------------------
// Stage 1: styles[b][i] = w[b]·affine_weight[i] / sqrt(512) + affine_bias[i]
// ---------------------------------------------------------------------------
__global__ __launch_bounds__(256) void k_styles(const float* __restrict__ w,
                                                const float* __restrict__ aw,
                                                const float* __restrict__ ab,
                                                float* __restrict__ styles) {
  int t = blockIdx.x * 256 + threadIdx.x;          // 16*512 threads
  int b = t >> 9, i = t & 511;
  float acc = 0.f;
#pragma unroll 8
  for (int j = 0; j < 512; ++j) acc += w[b * 512 + j] * aw[i * 512 + j];
  styles[t] = acc * 0.04419417382415922f + ab[i];  // 1/sqrt(512)
}

// ---------------------------------------------------------------------------
// Stage 2a: dcoefs[b][o] = rsqrt( sum_i (sum_9 weight^2) * styles^2 + 1e-8 )
// ---------------------------------------------------------------------------
__global__ __launch_bounds__(256) void k_dcoefs(const float* __restrict__ weight,
                                                const float* __restrict__ styles,
                                                float* __restrict__ dcoefs) {
  int t = blockIdx.x * 256 + threadIdx.x;          // 16*512 threads
  int b = t >> 9, o = t & 511;
  float acc = 0.f;
  for (int i = 0; i < 512; ++i) {
    const float* wp = weight + ((size_t)o * 512 + i) * 9;
    float ws = 0.f;
#pragma unroll
    for (int q = 0; q < 9; ++q) ws += wp[q] * wp[q];
    float s = styles[b * 512 + i];
    acc += ws * s * s;
  }
  dcoefs[t] = rsqrtf(acc + 1e-8f);
}

// ---------------------------------------------------------------------------
// Stage 2b: XS[b][pos][ci] = bf16( x[b][ci][pos] * styles[b][ci] )   (NHWC)
// A and B fragments both use contiguous K here, so no permutation needed.
// ---------------------------------------------------------------------------
__global__ __launch_bounds__(256) void k_pack_x(const float* __restrict__ x,
                                                const float* __restrict__ styles,
                                                uint16_t* __restrict__ XS) {
  int idx = blockIdx.x * 256 + threadIdx.x;        // 16*1024*512
  int ci  = idx & 511;
  int pos = (idx >> 9) & 1023;
  int b   = idx >> 19;
  float v = x[((size_t)(b * 512 + ci)) * 1024 + pos] * styles[b * 512 + ci];
  XS[idx] = f32_to_bf16_bits(v);
}

// ---------------------------------------------------------------------------
// Stage 2c: WT[t][co][ci] = bf16( weight[co][ci][t] )
// No K permutation: A and B fragments both use contiguous k (k = 16h + s),
// lane<16: K{0-15}, lane>=16: K{16-31}, so each fragment is a
// contiguous 16-element run at offset (hi_half ? 16 : 0).
// ---------------------------------------------------------------------------
__global__ __launch_bounds__(256) void k_pack_w(const float* __restrict__ weight,
                                                uint16_t* __restrict__ WT) {
  int idx = blockIdx.x * 256 + threadIdx.x;        // 9*512*512
  int ci = idx & 511;
  int co = (idx >> 9) & 511;
  int t  = idx >> 18;
  int j  = ci & 31;
  int pj = j;
  int dci = (ci & ~31) | pj;
  WT[((size_t)(t * 512 + co) << 9) + dci] =
      f32_to_bf16_bits(weight[((size_t)(co * 512 + ci)) * 9 + t]);
}

// ---------------------------------------------------------------------------
// Stage 3: phase-decomposed transposed conv as implicit GEMM on WMMA.
//   y1[b][co][oy][ox], oy,ox in [0,65).  Phase p=(py,px)=(oy&1,ox&1).
//   Taps: kh ≡ py (mod 2), kw ≡ px (mod 2); input pixel (sy+dy, sx+dx)
//   with dy=(py+kh)/2-1.  Flipped weight tap tf=(2-kh)*3+(2-kw).
//   WG = 8 waves = 128(M) x 64(N) tile; wave = 32x32 = 4 WMMA tiles.
//   Per tap, K is staged through LDS in KCH chunks (A: 64KB, B: 32KB),
//   halo positions zero-filled at stage time -> branch-free WMMA loop.
// ---------------------------------------------------------------------------
__global__ __launch_bounds__(256) void k_upconv_wmma(const uint16_t* __restrict__ WT,
                                                     const uint16_t* __restrict__ XS,
                                                     float* __restrict__ Y1) {
  extern __shared__ uint16_t smem[];
  uint16_t* lA = smem;                    // [128][KPAD]
  uint16_t* lB = smem + 128 * KPAD;       // [64][KPAD]

  const int tid  = threadIdx.x;
  const int lane = tid & 31;
  const int wv   = tid >> 5;              // 0..7
  const int b    = blockIdx.z;
  const int p    = blockIdx.y >> 2;       // phase 0..3  (py*2+px)
  const int mt   = blockIdx.y & 3;        // M tile (128 each)
  const int nt   = blockIdx.x;            // N tile (64 each)
  const int py = p >> 1, px = p & 1;
  const int Hp = py ? 32 : 33;
  const int Wp = px ? 32 : 33;
  const int Np = Hp * Wp;
  if (nt * 64 >= Np) return;

  const int mbase = mt * 128;
  const int nbase = nt * 64;
  const int wm = (wv & 3) * 32;           // wave M offset within tile
  const int wn = (wv >> 2) * 32;          // wave N offset within tile
  const bool hi_half = (lane >= 16);
  const int klo = hi_half ? 16 : 0;       // both frags: contiguous 16 K values

  v8f acc[2][2] = {};

  for (int kh = py; kh < 3; kh += 2) {
    const int dy = (py + kh) / 2 - 1;
    for (int kw = px; kw < 3; kw += 2) {
      const int dx = (px + kw) / 2 - 1;
      const int tf = (2 - kh) * 3 + (2 - kw);
      const uint16_t* Wtap = WT + (size_t)tf * COUT * CIN;

      for (int kc = 0; kc < CIN; kc += KCH) {
        __syncthreads();                       // previous chunk fully consumed
        // ---- stage A: 128 rows x KCH bf16 (always in-bounds) ----
#pragma unroll
        for (int c = tid; c < 128 * (KCH / 8); c += 256) {
          int row = c >> 5;                    // KCH/8 = 32 chunks per row
          int col = (c & 31) * 8;
          stage16(lA + row * KPAD + col,
                  Wtap + (size_t)(mbase + row) * CIN + kc + col);
        }
        // ---- stage B: 64 positions x KCH bf16 (zero-fill halo) ----
#pragma unroll
        for (int c = tid; c < 64 * (KCH / 8); c += 256) {
          int pos = c >> 5;
          int col = (c & 31) * 8;
          int s  = nbase + pos;
          int iy = s / Wp + dy;
          int ix = s % Wp + dx;
          uint16_t* dst = lB + pos * KPAD + col;
          if ((iy >= 0) & (iy < RIN) & (ix >= 0) & (ix < RIN)) {
            stage16(dst, XS + ((size_t)(b * 1024 + iy * RIN + ix)) * CIN + kc + col);
          } else {
            *(v8u16*)dst = (v8u16){0, 0, 0, 0, 0, 0, 0, 0};
          }
        }
        async_stage_fence();
        __syncthreads();

        // ---- compute: KCH/32 barrier-free WMMA k-steps ----
#pragma unroll 2
        for (int k = 0; k < KCH; k += 32) {
          const uint16_t* Ar0 = lA + (wm + (lane & 15)) * KPAD + k + klo;
          const uint16_t* Br0 = lB + (wn + (lane & 15)) * KPAD + k + klo;
          v16bf a0 = make_frag16(Ar0);
          v16bf a1 = make_frag16(Ar0 + 16 * KPAD);
          v16bf b0 = make_frag16(Br0);
          v16bf b1 = make_frag16(Br0 + 16 * KPAD);
          acc[0][0] = __builtin_amdgcn_wmma_f32_16x16x32_bf16(false, a0, false, b0,
                          (short)0, acc[0][0], false, false);
          acc[0][1] = __builtin_amdgcn_wmma_f32_16x16x32_bf16(false, a0, false, b1,
                          (short)0, acc[0][1], false, false);
          acc[1][0] = __builtin_amdgcn_wmma_f32_16x16x32_bf16(false, a1, false, b0,
                          (short)0, acc[1][0], false, false);
          acc[1][1] = __builtin_amdgcn_wmma_f32_16x16x32_bf16(false, a1, false, b1,
                          (short)0, acc[1][1], false, false);
        }
      }
    }
  }

  // Store: C/D layout — lane 0..15: N=lane, VGPR r -> M=r; lanes 16..31 -> M=r+8.
  const int m0 = mbase + wm;
  const int n0 = nbase + wn;
  const int nlane = lane & 15;
  const int mbump = hi_half ? 8 : 0;
#pragma unroll
  for (int am = 0; am < 2; ++am) {
#pragma unroll
    for (int an = 0; an < 2; ++an) {
      int s = n0 + an * 16 + nlane;
      if (s >= Np) continue;
      int sy = s / Wp, sx = s % Wp;
      int oy = 2 * sy + py, ox = 2 * sx + px;
      size_t base = ((size_t)(b * COUT + m0 + am * 16 + mbump)) * Y1PIX
                  + (size_t)oy * Y1DIM + ox;
#pragma unroll
      for (int r = 0; r < 8; ++r)
        Y1[base + (size_t)r * Y1PIX] = acc[am][an][r];
    }
  }
}

// ---------------------------------------------------------------------------
// Stage 4: 4x4 bilinear blur (pad 1) + noise + demod + bias + gained leaky.
// ---------------------------------------------------------------------------
__global__ __launch_bounds__(256) void k_blur_epilogue(const float* __restrict__ Y1,
                                                       const float* __restrict__ noise,
                                                       const float* __restrict__ dcoefs,
                                                       const float* __restrict__ bias,
                                                       const float* __restrict__ nstr,
                                                       float* __restrict__ out) {
  int idx = blockIdx.x * 256 + threadIdx.x;        // 16*512*64*64
  int x  = idx & 63;
  int y  = (idx >> 6) & 63;
  int co = (idx >> 12) & 511;
  int b  = idx >> 21;
  const float h[4] = {1.f, 3.f, 3.f, 1.f};
  const float* src = Y1 + ((size_t)(b * COUT + co)) * Y1PIX;
  float s = 0.f;
#pragma unroll
  for (int u = 0; u < 4; ++u) {
    int yy = y + u - 1;
    if (yy < 0 || yy >= Y1DIM) continue;
    float rs = 0.f;
#pragma unroll
    for (int v = 0; v < 4; ++v) {
      int xx = x + v - 1;
      if (xx < 0 || xx >= Y1DIM) continue;
      rs += h[v] * src[(size_t)yy * Y1DIM + xx];
    }
    s += h[u] * rs;
  }
  s *= (1.f / 16.f);                                // h⊗h / 64 * 4
  float val = noise[(size_t)b * 4096 + y * 64 + x] * nstr[0]
            + s * dcoefs[b * 512 + co];
  val = (val + bias[co]) * 1.41421356237309515f;    // sqrt(2) gain
  out[idx] = (val >= 0.f) ? val : 0.2f * val;
}

// ---------------------------------------------------------------------------
extern "C" void kernel_launch(void* const* d_in, const int* in_sizes, int n_in,
                              void* d_out, int out_size, void* d_ws, size_t ws_size,
                              hipStream_t stream) {
  (void)in_sizes; (void)n_in; (void)out_size; (void)ws_size;
  const float* x      = (const float*)d_in[0];   // (16,512,32,32)
  const float* w      = (const float*)d_in[1];   // (16,512)
  const float* noise  = (const float*)d_in[2];   // (16,1,64,64)
  const float* aw     = (const float*)d_in[3];   // (512,512)
  const float* ab     = (const float*)d_in[4];   // (512,)
  const float* weight = (const float*)d_in[5];   // (512,512,3,3)
  const float* bias   = (const float*)d_in[6];   // (512,)
  const float* nstr   = (const float*)d_in[7];   // scalar
  float* out = (float*)d_out;

  char* ws = (char*)d_ws;
  float*    styles = (float*)(ws);                               // 32 KB
  float*    dcoefs = (float*)(ws + 32768);                       // 32 KB
  uint16_t* XS     = (uint16_t*)(ws + 65536);                    // 16 MB
  uint16_t* WT     = (uint16_t*)(ws + 65536 + 16777216);         // 4.5 MB
  float*    Y1     = (float*)(ws + 65536 + 16777216 + 4718592);  // 138.5 MB

  // styles -> (dcoefs | XS), WT independent, then GEMM, then epilogue.
  k_styles<<<dim3(32), dim3(256), 0, stream>>>(w, aw, ab, styles);
  k_dcoefs<<<dim3(32), dim3(256), 0, stream>>>(weight, styles, dcoefs);
  k_pack_x<<<dim3((NB * 1024 * CIN) / 256), dim3(256), 0, stream>>>(x, styles, XS);
  k_pack_w<<<dim3((9 * COUT * CIN) / 256), dim3(256), 0, stream>>>(weight, WT);
  k_upconv_wmma<<<dim3(18, 16, NB), dim3(256), LDS_BYTES, stream>>>(WT, XS, Y1);
  k_blur_epilogue<<<dim3((NB * COUT * ROUT * ROUT) / 256), dim3(256), 0, stream>>>(
      Y1, noise, dcoefs, bias, nstr, out);
}
